// VaeDecoder_18691697672723
// MI455X (gfx1250) — matrix operand order repaired and checked by reading, not compiled, not run on hardware
//
#include <hip/hip_runtime.h>
#include <hip/hip_bf16.h>
#include <math.h>

typedef __attribute__((ext_vector_type(16))) _Float16 v16h;
typedef __attribute__((ext_vector_type(8)))  _Float16 v8h;
typedef __attribute__((ext_vector_type(4)))  _Float16 v4h;
typedef __attribute__((ext_vector_type(8)))  float    v8f;
typedef __attribute__((ext_vector_type(4)))  float    v4f;

#define LATD 100
#define TSTEPS 30
#define DTC 0.03f
#define LOG2E 1.44269504089f
#define INV2PI 0.15915494309f

// f16 workspace layout (element offsets within half region)
#define HO_WHH   0        // [256][64]
#define HO_WINIT 16384    // [64][128]  (K padded 100->128)
#define HO_WL0   24576    // [64][128]
#define HO_WL1   32768    // [64][64]
#define HO_WL2   36864    // [64][64]
#define HO_WL3   40960    // [16][64]   (N padded 2->16)
#define HO_WH2C  41984    // [16][64]   (N padded 2->16)
#define H_TOTAL  43008

__device__ __forceinline__ void lds_fence() {
  asm volatile("s_wait_dscnt 0x0" ::: "memory");
}

__device__ __forceinline__ v8f wmma_f16(v16h a, v16h b, v8f c) {
  return __builtin_amdgcn_wmma_f32_16x16x32_f16(false, a, false, b, (short)0, c,
                                                false, false);
}

// branch-free fast math on the native VALU transcendental pipe
__device__ __forceinline__ float frcp(float x) { return __builtin_amdgcn_rcpf(x); }
__device__ __forceinline__ float fexp(float x) { return __builtin_amdgcn_exp2f(x * LOG2E); }
__device__ __forceinline__ float sigm(float x) { return frcp(1.f + fexp(-x)); }
__device__ __forceinline__ float ftanh(float x) { return 2.f * frcp(1.f + fexp(-2.f * x)) - 1.f; }
__device__ __forceinline__ float fsin(float x) { return __builtin_amdgcn_sinf(x * INV2PI); }
__device__ __forceinline__ float fcos(float x) { return __builtin_amdgcn_cosf(x * INV2PI); }
__device__ __forceinline__ float lrelu(float x) { return x > 0.f ? x : 0.01f * x; }

// A fragment (16 rows x 32 K) from f16 row-major [16][ld] (LDS), vector loads.
// A layout: lane L -> row m=L&15; elems 0..7 = K kb..kb+7, 8..15 = K kb+16..kb+23, kb=(L>=16)*8.
__device__ __forceinline__ v16h fragA_h(const _Float16* base, int ld, int k0, int lane) {
  const int m  = lane & 15;
  const int kb = k0 + ((lane >> 4) << 3);
  const _Float16* p = base + m * ld + kb;
  v8h lo = *(const v8h*)p;
  v8h hi = *(const v8h*)(p + 16);
  v16h r;
#pragma unroll
  for (int i = 0; i < 8; ++i) { r[i] = lo[i]; r[i + 8] = hi[i]; }
  return r;
}

// B fragment (32 K x 16 cols) from f16 row-major weights [N][ldk] (global ws or LDS).
// B layout: lane L -> col n=L&15; elems 0..7 = K kb..kb+7, 8..15 = K kb+16..kb+23, kb=k0+(L>=16)*8.
__device__ __forceinline__ v16h fragB_h(const _Float16* W, int ldk, int nBase,
                                        int k0, int lane) {
  const int n  = nBase + (lane & 15);
  const int ks = k0 + ((lane >> 4) << 3);
  const _Float16* p = W + n * ldk + ks;
  v8h lo = *(const v8h*)p;
  v8h hi = *(const v8h*)(p + 16);
  v16h r;
#pragma unroll
  for (int i = 0; i < 8; ++i) { r[i] = lo[i]; r[i + 8] = hi[i]; }
  return r;
}

// Fold W_emb into W_ih:  Wc = W_ih @ W_emb (256x4),  bc = W_ih@b_emb + b_ih + b_hh
__global__ void wc_kernel(const float* __restrict__ W_emb, const float* __restrict__ b_emb,
                          const float* __restrict__ W_ih, const float* __restrict__ b_ih,
                          const float* __restrict__ b_hh,
                          float* __restrict__ wc, float* __restrict__ bc) {
  int n = blockIdx.x * blockDim.x + threadIdx.x;
  if (n < 256) {
    float a0 = 0.f, a1 = 0.f, a2 = 0.f, a3 = 0.f, bb = 0.f;
    for (int k = 0; k < 64; ++k) {
      float w = W_ih[n * 64 + k];
      bb += w * b_emb[k];
      a0 += w * W_emb[k * 4 + 0];
      a1 += w * W_emb[k * 4 + 1];
      a2 += w * W_emb[k * 4 + 2];
      a3 += w * W_emb[k * 4 + 3];
    }
    wc[n * 4 + 0] = a0; wc[n * 4 + 1] = a1;
    wc[n * 4 + 2] = a2; wc[n * 4 + 3] = a3;
    bc[n] = bb + b_ih[n] + b_hh[n];
  }
}

// Pre-pad + f32->f16 convert all weight matrices into workspace.
__global__ void prep_kernel(const float* __restrict__ W_hh, const float* __restrict__ W_init,
                            const float* __restrict__ Wl0, const float* __restrict__ Wl1,
                            const float* __restrict__ Wl2, const float* __restrict__ Wl3,
                            const float* __restrict__ W_h2c, _Float16* __restrict__ hws) {
  int idx = blockIdx.x * blockDim.x + threadIdx.x;
  if (idx < 16384) { hws[HO_WHH + idx] = (_Float16)W_hh[idx]; return; }
  idx -= 16384;
  if (idx < 8192) {
    int n = idx >> 7, k = idx & 127;
    hws[HO_WINIT + idx] = (k < LATD) ? (_Float16)W_init[n * LATD + k] : (_Float16)0.f;
    return;
  }
  idx -= 8192;
  if (idx < 8192) {
    int n = idx >> 7, k = idx & 127;
    hws[HO_WL0 + idx] = (k < LATD) ? (_Float16)Wl0[n * LATD + k] : (_Float16)0.f;
    return;
  }
  idx -= 8192;
  if (idx < 4096) { hws[HO_WL1 + idx] = (_Float16)Wl1[idx]; return; }
  idx -= 4096;
  if (idx < 4096) { hws[HO_WL2 + idx] = (_Float16)Wl2[idx]; return; }
  idx -= 4096;
  if (idx < 1024) {
    int n = idx >> 6;
    hws[HO_WL3 + idx] = (n < 2) ? (_Float16)Wl3[idx] : (_Float16)0.f;
    return;
  }
  idx -= 1024;
  if (idx < 1024) {
    int n = idx >> 6;
    hws[HO_WH2C + idx] = (n < 2) ? (_Float16)W_h2c[idx] : (_Float16)0.f;
  }
}

__global__ __launch_bounds__(128)
void decoder_kernel(const float* __restrict__ z, const float* __restrict__ init_state,
                    const float* __restrict__ b_h2c, const float* __restrict__ b_init,
                    const float* __restrict__ bl0, const float* __restrict__ bl1,
                    const float* __restrict__ bl2, const float* __restrict__ bl3,
                    const float* __restrict__ Wc_g, const float* __restrict__ bc_g,
                    const _Float16* __restrict__ hws,
                    float* __restrict__ out, int Btot) {
  __shared__ _Float16 sW_hh[256 * 64];     // 32 KB, hot in T-loop
  __shared__ _Float16 sWh2c[16 * 64];      //  2 KB, hot in T-loop
  __shared__ _Float16 hbuf[4][2048];       // 16 KB: per-wave z-staging / h / mlp ping
  __shared__ float    ctl[4][16][2];       // control exchange

  const int tid = threadIdx.x;
  for (int i = tid * 8; i < 256 * 64; i += blockDim.x * 8)
    *(v8h*)(sW_hh + i) = *(const v8h*)(hws + HO_WHH + i);
  for (int i = tid * 8; i < 16 * 64; i += blockDim.x * 8)
    *(v8h*)(sWh2c + i) = *(const v8h*)(hws + HO_WH2C + i);
  __syncthreads();

  const int wave = tid >> 5;
  const int lane = tid & 31;
  const int rowBase = blockIdx.x * 64 + wave * 16;
  const int nLow = lane & 15;
  const int mOff = (lane >> 4) << 3;

  _Float16* zb     = hbuf[wave];          // 16 x 128 staging
  _Float16* h0buf  = hbuf[wave];          // 16 x 64 (first half)
  _Float16* mlpbuf = hbuf[wave] + 1024;   // 16 x 64 (second half)

  // ---- stage this wave's 16 z rows into LDS, coalesced, guard-free ----
  {
    const float* zrow = z + (long)rowBase * LATD;
    for (int idx = lane; idx < 16 * LATD; idx += 32) {
      int r = idx / LATD, k = idx - r * LATD;
      zb[r * 128 + k] = (_Float16)zrow[idx];
    }
    for (int idx = lane; idx < 16 * 28; idx += 32) {  // zero pad cols 100..127
      int r = idx / 28, k = idx - r * 28;
      zb[r * 128 + LATD + k] = (_Float16)0.f;
    }
  }
  lds_fence();

  v16h za[4];
#pragma unroll
  for (int c = 0; c < 4; ++c) za[c] = fragA_h(zb, 128, 32 * c, lane);
  lds_fence();

  // ---- h0 = z @ W_init^T + b_init ; c0 = h0 ----
  v8f cst[4];
#pragma unroll
  for (int j = 0; j < 4; ++j) {
    float bv = b_init[16 * j + nLow];
    v8f acc;
#pragma unroll
    for (int r = 0; r < 8; ++r) acc[r] = bv;
#pragma unroll
    for (int c = 0; c < 4; ++c)
      acc = wmma_f16(za[c], fragB_h(hws + HO_WINIT, 128, 16 * j, 32 * c, lane), acc);
    cst[j] = acc;
#pragma unroll
    for (int r = 0; r < 8; ++r)
      h0buf[(r + mOff) * 64 + 16 * j + nLow] = (_Float16)acc[r];
  }

  // ---- label head ----
#pragma unroll
  for (int j = 0; j < 4; ++j) {
    float bv = bl0[16 * j + nLow];
    v8f acc;
#pragma unroll
    for (int r = 0; r < 8; ++r) acc[r] = bv;
#pragma unroll
    for (int c = 0; c < 4; ++c)
      acc = wmma_f16(za[c], fragB_h(hws + HO_WL0, 128, 16 * j, 32 * c, lane), acc);
#pragma unroll
    for (int r = 0; r < 8; ++r)
      mlpbuf[(r + mOff) * 64 + 16 * j + nLow] = (_Float16)lrelu(acc[r]);
  }
  lds_fence();

  {
    const _Float16* Wls[2] = {hws + HO_WL1, hws + HO_WL2};
    const float* bls[2] = {bl1, bl2};
#pragma unroll
    for (int L = 0; L < 2; ++L) {
      v16h a0 = fragA_h(mlpbuf, 64, 0, lane);
      v16h a1 = fragA_h(mlpbuf, 64, 32, lane);
      lds_fence();
#pragma unroll
      for (int j = 0; j < 4; ++j) {
        float bv = bls[L][16 * j + nLow];
        v8f acc;
#pragma unroll
        for (int r = 0; r < 8; ++r) acc[r] = bv;
        acc = wmma_f16(a0, fragB_h(Wls[L], 64, 16 * j, 0, lane), acc);
        acc = wmma_f16(a1, fragB_h(Wls[L], 64, 16 * j, 32, lane), acc);
#pragma unroll
        for (int r = 0; r < 8; ++r)
          mlpbuf[(r + mOff) * 64 + 16 * j + nLow] = (_Float16)lrelu(acc[r]);
      }
      lds_fence();
    }
  }

  {  // layer 3 (N padded to 16)
    v16h a0 = fragA_h(mlpbuf, 64, 0, lane);
    v16h a1 = fragA_h(mlpbuf, 64, 32, lane);
    float bv = 0.f;
    if (nLow < 2) bv = bl3[nLow];
    v8f acc;
#pragma unroll
    for (int r = 0; r < 8; ++r) acc[r] = bv;
    acc = wmma_f16(a0, fragB_h(hws + HO_WL3, 64, 0, 0, lane), acc);
    acc = wmma_f16(a1, fragB_h(hws + HO_WL3, 64, 0, 32, lane), acc);
    if (nLow < 2) {
      long base = (long)Btot * TSTEPS * 4;
#pragma unroll
      for (int r = 0; r < 8; ++r) {
        int row = rowBase + r + mOff;
        out[base + (long)row * 2 + nLow] = lrelu(acc[r]);
      }
    }
  }

  // ---- hoist T-loop invariants into VGPRs ----
  v4h  wcv[16];   // state->gate weights for this lane's column (lanes 0..15 live)
  float bcv[16];  // combined gate bias for this lane's column
#pragma unroll
  for (int tl = 0; tl < 16; ++tl) {
    int nB = tl * 16;
    bcv[tl] = bc_g[nB + nLow];
    v4h w;
#pragma unroll
    for (int i = 0; i < 4; ++i) w[i] = (_Float16)0.f;
    if (lane < 16) {
      v4f wf = *(const v4f*)(Wc_g + (nB + nLow) * 4);
      w[0] = (_Float16)wf[0]; w[1] = (_Float16)wf[1];
      w[2] = (_Float16)wf[2]; w[3] = (_Float16)wf[3];
    }
    wcv[tl] = w;
  }
  float cbias = 0.f;
  if (nLow < 2) cbias = b_h2c[nLow];

  // ---- per-lane plant state (lanes 0..15 own rows) ----
  float sx = 0.f, sy = 0.f, spsi = 0.f, sv = 0.f;
  if (lane < 16) {
    const float* ip = init_state + (long)(rowBase + lane) * 4;
    sx = ip[0]; sy = ip[1]; spsi = ip[2]; sv = ip[3];
  }

  lds_fence();
  v16h ha0 = fragA_h(h0buf, 64, 0, lane);
  v16h ha1 = fragA_h(h0buf, 64, 32, lane);
  lds_fence();

  // state A-fragment: zero once; only elems 0..3 (lanes<16) change per step
  v16h as;
#pragma unroll
  for (int i = 0; i < 16; ++i) as[i] = (_Float16)0.f;

  for (int t = 0; t < TSTEPS; ++t) {
    if (lane < 16) {
      as[0] = (_Float16)sx; as[1] = (_Float16)sy;
      as[2] = (_Float16)spsi; as[3] = (_Float16)sv;
    }

#pragma unroll
    for (int j = 0; j < 4; ++j) {
      v8f g4[4];
#pragma unroll
      for (int gi = 0; gi < 4; ++gi) {
        const int nB = gi * 64 + j * 16;
        const int tl = gi * 4 + j;
        v8f acc;
#pragma unroll
        for (int r = 0; r < 8; ++r) acc[r] = bcv[tl];
        acc = wmma_f16(ha0, fragB_h(sW_hh, 64, nB, 0, lane), acc);
        acc = wmma_f16(ha1, fragB_h(sW_hh, 64, nB, 32, lane), acc);
        v16h bs;
#pragma unroll
        for (int i = 0; i < 16; ++i) bs[i] = (_Float16)0.f;
        bs[0] = wcv[tl][0]; bs[1] = wcv[tl][1];
        bs[2] = wcv[tl][2]; bs[3] = wcv[tl][3];
        acc = wmma_f16(as, bs, acc);
        g4[gi] = acc;
      }
#pragma unroll
      for (int r = 0; r < 8; ++r) {
        float ig = sigm(g4[0][r]);
        float fg = sigm(g4[1][r]);
        float gg = ftanh(g4[2][r]);
        float og = sigm(g4[3][r]);
        float cv = fg * cst[j][r] + ig * gg;
        cst[j][r] = cv;
        float hv = og * ftanh(cv);
        h0buf[(r + mOff) * 64 + j * 16 + nLow] = (_Float16)hv;
      }
    }
    lds_fence();

    // new-h A fragments: feed control tile now, gates next step
    v16h hn0 = fragA_h(h0buf, 64, 0, lane);
    v16h hn1 = fragA_h(h0buf, 64, 32, lane);

    v8f acc2;
#pragma unroll
    for (int r = 0; r < 8; ++r) acc2[r] = cbias;
    acc2 = wmma_f16(hn0, fragB_h(sWh2c, 64, 0, 0, lane), acc2);
    acc2 = wmma_f16(hn1, fragB_h(sWh2c, 64, 0, 32, lane), acc2);
    if (nLow < 2) {
#pragma unroll
      for (int r = 0; r < 8; ++r) ctl[wave][r + mOff][nLow] = acc2[r];
    }
    lds_fence();

    if (lane < 16) {
      float pedal = ctl[wave][lane][0];
      float steer = ctl[wave][lane][1];
      float beta = fminf(fmaxf(steer, -0.5f), 0.5f);
      float v1 = fminf(fmaxf(sv + pedal * DTC, 0.f), 10.f);
      float tanb = fsin(beta) * frcp(fcos(beta));
      float psid = fminf(fmaxf(sv * tanb * 0.4f, -1.57f), 1.57f);
      float psi1 = psid * DTC + spsi;
      sx = v1 * fcos(psi1) * DTC + sx;
      sy = v1 * fsin(psi1) * DTC + sy;
      spsi = psi1; sv = v1;
      float* op = out + ((long)(rowBase + lane) * TSTEPS + t) * 4;
      op[0] = sx; op[1] = sy; op[2] = spsi; op[3] = sv;
    }
    ha0 = hn0; ha1 = hn1;
  }
}

extern "C" void kernel_launch(void* const* d_in, const int* in_sizes, int n_in,
                              void* d_out, int out_size, void* d_ws, size_t ws_size,
                              hipStream_t stream) {
  const float* z      = (const float*)d_in[0];
  const float* init_s = (const float*)d_in[1];
  const float* W_emb  = (const float*)d_in[2];
  const float* b_emb  = (const float*)d_in[3];
  const float* W_ih   = (const float*)d_in[4];
  const float* b_ih   = (const float*)d_in[5];
  const float* W_hh   = (const float*)d_in[6];
  const float* b_hh   = (const float*)d_in[7];
  const float* W_h2c  = (const float*)d_in[8];
  const float* b_h2c  = (const float*)d_in[9];
  const float* W_init = (const float*)d_in[10];
  const float* b_init = (const float*)d_in[11];
  const float* Wl0    = (const float*)d_in[12];
  const float* bl0    = (const float*)d_in[13];
  const float* Wl1    = (const float*)d_in[14];
  const float* bl1    = (const float*)d_in[15];
  const float* Wl2    = (const float*)d_in[16];
  const float* bl2    = (const float*)d_in[17];
  const float* Wl3    = (const float*)d_in[18];
  const float* bl3    = (const float*)d_in[19];
  float* out = (float*)d_out;
  const int Btot = in_sizes[1] / 4;

  float* wc = (float*)d_ws;                       // 1024 floats
  float* bc = wc + 1024;                          // 256 floats
  _Float16* hws = (_Float16*)(wc + 1280);         // f16 weight region

  wc_kernel<<<1, 256, 0, stream>>>(W_emb, b_emb, W_ih, b_ih, b_hh, wc, bc);
  prep_kernel<<<(H_TOTAL + 255) / 256, 256, 0, stream>>>(
      W_hh, W_init, Wl0, Wl1, Wl2, Wl3, W_h2c, hws);
  decoder_kernel<<<Btot / 64, 128, 0, stream>>>(
      z, init_s, b_h2c, b_init, bl0, bl1, bl2, bl3, wc, bc, hws, out, Btot);
}
